// MultiHeadAttention_74500502716958
// MI455X (gfx1250) — matrix operand order repaired and checked by reading, not compiled, not run on hardware
//
#include <hip/hip_runtime.h>

#define HID   1024
#define NH    16
#define HD    64
#define SEQ   1024
#define BATCH 4

typedef __attribute__((ext_vector_type(16))) __bf16 bhalf16;
typedef __attribute__((ext_vector_type(8)))  float  float8;
typedef int v4i_ __attribute__((vector_size(16)));

union AF { bhalf16 v; uint4 q[2]; };

#if defined(__HIP_DEVICE_COMPILE__) && __has_builtin(__builtin_amdgcn_global_load_async_to_lds_b128)
#define ASYNC_LDS 1
#endif

// Stage 16 bytes global -> LDS. Async (ASYNCcnt) path when available,
// otherwise bounce through VGPRs.
__device__ __forceinline__ void stage16(void* lds_dst, const void* gsrc) {
#ifdef ASYNC_LDS
  __builtin_amdgcn_global_load_async_to_lds_b128(
      (v4i_*)const_cast<void*>(gsrc), (v4i_*)lds_dst, 0, 0);
#else
  *(uint4*)lds_dst = *(const uint4*)gsrc;
#endif
}

__device__ __forceinline__ void wait_stage() {
#ifdef ASYNC_LDS
#if __has_builtin(__builtin_amdgcn_s_wait_asynccnt)
  __builtin_amdgcn_s_wait_asynccnt(0);
#else
  asm volatile("s_wait_asynccnt 0" ::: "memory");
#endif
#endif
}

__device__ __forceinline__ __bf16 tobf(float f) {
  unsigned u = __builtin_bit_cast(unsigned, f);
  u = (u + 0x7FFFu + ((u >> 16) & 1u)) >> 16;   // round-to-nearest-even
  unsigned short s = (unsigned short)u;
  return __builtin_bit_cast(__bf16, s);
}

// ---------------- conversion kernels ----------------

__global__ __launch_bounds__(256) void cvt_bf16(const float* __restrict__ in,
                                                __bf16* __restrict__ out, int n) {
  int i = (blockIdx.x * 256 + threadIdx.x) * 4;
  if (i + 3 < n) {
    float4 f = *(const float4*)(in + i);
    out[i + 0] = tobf(f.x);
    out[i + 1] = tobf(f.y);
    out[i + 2] = tobf(f.z);
    out[i + 3] = tobf(f.w);
  }
}

// out[n][k] = in[k][n]  (transpose so W columns become contiguous K runs)
__global__ __launch_bounds__(256) void cvt_bf16_t(const float* __restrict__ in,
                                                  __bf16* __restrict__ out) {
  int idx = blockIdx.x * 256 + threadIdx.x;      // idx = n*1024 + k
  int nrow = idx >> 10, k = idx & 1023;
  out[idx] = tobf(in[k * HID + nrow]);
}

// ---------------- bf16 WMMA GEMM: C = A @ Wt^T + bias ----------------
// A  : [M][1024] bf16 row-major (M = 4096)
// Wt : [1024][1024] bf16, Wt[n][k]
// MODE 0: f32 out [M][1024]
// MODE 1: bf16 out [b,h,t,d]   (Q / K)
// MODE 2: bf16 out [b,h,d,t]   (V, pre-transposed for PV WMMA)
template<int MODE>
__global__ __launch_bounds__(256) void gemm_bf16(
    const __bf16* __restrict__ A,
    const __bf16* __restrict__ Wt,
    const float*  __restrict__ bias,
    void* __restrict__ dstv)
{
  __shared__ __align__(16) __bf16 As[128 * 32];   // [row][k]
  __shared__ __align__(16) __bf16 Bs[128 * 32];   // [col][k]
  const int tid      = threadIdx.x;
  const int blockRow = blockIdx.y * 128;
  const int blockCol = blockIdx.x * 128;
  const int w = tid >> 5, lane = tid & 31, laneN = lane & 15, kg = lane >> 4;

  float8 acc[8];
#pragma unroll
  for (int i = 0; i < 8; ++i) acc[i] = (float8)0.0f;

  for (int kk = 0; kk < HID; kk += 32) {
    __syncthreads();
#pragma unroll
    for (int i = 0; i < 2; ++i) {
      int c = tid + i * 256;          // 512 16-byte chunks each for A and B
      int row = c >> 2, kq = c & 3;
      stage16(&As[row * 32 + kq * 8],
              &A[(size_t)(blockRow + row) * HID + kk + kq * 8]);
      stage16(&Bs[row * 32 + kq * 8],
              &Wt[(size_t)(blockCol + row) * HID + kk + kq * 8]);
    }
    wait_stage();
    __syncthreads();

    AF a;   // ISA 16-bit A layout: lane=row, kg picks K 0-7/16-23 vs 8-15/24-31
    const __bf16* ap = &As[(w * 16 + laneN) * 32];
    a.q[0] = *(const uint4*)(ap + kg * 8);
    a.q[1] = *(const uint4*)(ap + 16 + kg * 8);
#pragma unroll
    for (int nt = 0; nt < 8; ++nt) {
      AF bf;  // B layout: lane=col, kg picks K 0-7/16-23 vs 8-15/24-31
      const __bf16* bp = &Bs[(nt * 16 + laneN) * 32];
      bf.q[0] = *(const uint4*)(bp + kg * 8);
      bf.q[1] = *(const uint4*)(bp + 16 + kg * 8);
      acc[nt] = __builtin_amdgcn_wmma_f32_16x16x32_bf16(
          false, a.v, false, bf.v, (short)0, acc[nt], false, false);
    }
  }

#pragma unroll
  for (int nt = 0; nt < 8; ++nt) {
    int gn = blockCol + nt * 16 + laneN;
    float bv = bias[gn];
#pragma unroll
    for (int r = 0; r < 8; ++r) {
      int gm = blockRow + w * 16 + r + kg * 8;   // C layout: lanes16-31 => M+8
      float val = acc[nt][r] + bv;
      if (MODE == 0) {
        ((float*)dstv)[(size_t)gm * HID + gn] = val;
      } else {
        int b = gm >> 10, t = gm & 1023;
        int head = gn >> 6, d = gn & 63;
        __bf16* dst = (__bf16*)dstv;
        if (MODE == 1) dst[(((size_t)(b * NH + head)) * SEQ + t) * HD + d] = tobf(val);
        else           dst[(((size_t)(b * NH + head)) * HD + d) * SEQ + t] = tobf(val);
      }
    }
  }
}

// ---------------- flash-style attention (one 128-query block per WG) --------
__global__ __launch_bounds__(256) void attn(
    const __bf16* __restrict__ Q,     // [B*NH][SEQ][HD]
    const __bf16* __restrict__ Km,    // [B*NH][SEQ][HD]
    const __bf16* __restrict__ V,     // [B*NH][HD][SEQ]
    const float*  __restrict__ mask,  // [B][SEQ]
    __bf16* __restrict__ ctx)         // [B][SEQ][HID]
{
  __shared__ __align__(16) __bf16 Ks[32 * 64];       // [key][d]
  __shared__ __align__(16) __bf16 Vs[64 * 32];       // [d][key]
  __shared__ __align__(16) __bf16 Ps[8 * 16 * 32];   // per-wave P scratch

  const int tid  = threadIdx.x;
  const int blk  = blockIdx.x;               // B*NH*8 = 512 blocks
  const int b    = blk >> 7;
  const int head = (blk >> 3) & 15;
  const int qblk = blk & 7;
  const int w = tid >> 5, lane = tid & 31, laneN = lane & 15, kg = lane >> 4;
  const size_t bh = (size_t)(b * NH + head);

  // Q fragments (16 rows x 64 head-dim = two K-steps), kept in registers
  AF qf0, qf1;
  {
    const __bf16* qp = Q + (bh * SEQ + (qblk * 128 + w * 16 + laneN)) * HD;
    qf0.q[0] = *(const uint4*)(qp + kg * 8);
    qf0.q[1] = *(const uint4*)(qp + 16 + kg * 8);
    qf1.q[0] = *(const uint4*)(qp + 32 + kg * 8);
    qf1.q[1] = *(const uint4*)(qp + 48 + kg * 8);
  }

  float8 o[4];
#pragma unroll
  for (int i = 0; i < 4; ++i) o[i] = (float8)0.0f;
  float mrow[8], lrow[8];
#pragma unroll
  for (int r = 0; r < 8; ++r) { mrow[r] = -1e30f; lrow[r] = 0.0f; }

  const float scale  = 0.125f;               // 1/sqrt(64)
  const float NEGINF = -4294967296.0f;       // -2^32

  for (int kb = 0; kb < SEQ; kb += 32) {
    __syncthreads();
    {
      int key = tid >> 3, ch = tid & 7;      // 32 keys x 8 chunks of 16B
      stage16(&Ks[key * 64 + ch * 8],
              &Km[(bh * SEQ + kb + key) * HD + ch * 8]);
      int d = tid >> 2, ch2 = tid & 3;       // 64 d-rows x 4 chunks
      stage16(&Vs[d * 32 + ch2 * 8],
              &V[(bh * HD + d) * SEQ + kb + ch2 * 8]);
    }
    if (kb + 32 < SEQ) {                     // global_prefetch_b8 for next tile
      __builtin_prefetch(&Km[(bh * SEQ + kb + 32 + (tid >> 3)) * HD], 0, 0);
      __builtin_prefetch(&V[(bh * HD + (tid >> 2)) * SEQ + kb + 32], 0, 0);
    }
    wait_stage();
    __syncthreads();

    // S = Q @ K^T for 16 queries x 32 keys (2 N-tiles x 2 K-steps = 4 WMMA)
    float8 s0 = (float8)0.0f, s1 = (float8)0.0f;
#pragma unroll
    for (int ks = 0; ks < 2; ++ks) {
      AF bk0, bk1;
      const __bf16* kp0 = &Ks[(laneN) * 64 + ks * 32];
      bk0.q[0] = *(const uint4*)(kp0 + kg * 8);
      bk0.q[1] = *(const uint4*)(kp0 + 16 + kg * 8);
      const __bf16* kp1 = &Ks[(16 + laneN) * 64 + ks * 32];
      bk1.q[0] = *(const uint4*)(kp1 + kg * 8);
      bk1.q[1] = *(const uint4*)(kp1 + 16 + kg * 8);
      const AF& aq = ks ? qf1 : qf0;
      s0 = __builtin_amdgcn_wmma_f32_16x16x32_bf16(false, aq.v, false, bk0.v,
                                                   (short)0, s0, false, false);
      s1 = __builtin_amdgcn_wmma_f32_16x16x32_bf16(false, aq.v, false, bk1.v,
                                                   (short)0, s1, false, false);
    }

    // online softmax (rows live in VGPR index r, cols across 16-lane groups)
    float m0 = mask[b * SEQ + kb + laneN] * NEGINF;
    float m1 = mask[b * SEQ + kb + 16 + laneN] * NEGINF;
    float al[8];
#pragma unroll
    for (int r = 0; r < 8; ++r) {
      float a0 = s0[r] * scale + m0;
      float a1 = s1[r] * scale + m1;
      float cm = fmaxf(a0, a1);
#pragma unroll
      for (int mx = 1; mx <= 8; mx <<= 1) cm = fmaxf(cm, __shfl_xor(cm, mx, 32));
      float mn = fmaxf(mrow[r], cm);
      al[r] = __expf(mrow[r] - mn);
      mrow[r] = mn;
      float p0 = __expf(a0 - mn), p1 = __expf(a1 - mn);
      float rsum = p0 + p1;
#pragma unroll
      for (int mx = 1; mx <= 8; mx <<= 1) rsum += __shfl_xor(rsum, mx, 32);
      lrow[r] = lrow[r] * al[r] + rsum;
      s0[r] = p0; s1[r] = p1;
    }
#pragma unroll
    for (int nt = 0; nt < 4; ++nt)
#pragma unroll
      for (int r = 0; r < 8; ++r) o[nt][r] *= al[r];

    // P: C-layout -> A-layout via per-wave LDS scratch
    __bf16* pw = &Ps[w * 512];
#pragma unroll
    for (int r = 0; r < 8; ++r) {
      int m = r + kg * 8;
      pw[m * 32 + laneN]      = tobf(s0[r]);
      pw[m * 32 + 16 + laneN] = tobf(s1[r]);
    }
    asm volatile("s_wait_dscnt 0" ::: "memory");
    AF pa;
    pa.q[0] = *(const uint4*)(&pw[laneN * 32 + kg * 8]);
    pa.q[1] = *(const uint4*)(&pw[laneN * 32 + 16 + kg * 8]);

    // O += P @ V  (4 N-tiles over head dim)
#pragma unroll
    for (int nt = 0; nt < 4; ++nt) {
      AF bv;
      const __bf16* vp = &Vs[(nt * 16 + laneN) * 32];
      bv.q[0] = *(const uint4*)(vp + kg * 8);
      bv.q[1] = *(const uint4*)(vp + 16 + kg * 8);
      o[nt] = __builtin_amdgcn_wmma_f32_16x16x32_bf16(false, pa.v, false, bv.v,
                                                      (short)0, o[nt], false, false);
    }
  }

  // normalize and store context bf16 [b][t][head*64+d]
#pragma unroll
  for (int nt = 0; nt < 4; ++nt) {
#pragma unroll
    for (int r = 0; r < 8; ++r) {
      int t = qblk * 128 + w * 16 + r + kg * 8;
      float val = o[nt][r] / lrow[r];
      ctx[((size_t)(b * SEQ + t)) * HID + head * HD + nt * 16 + laneN] = tobf(val);
    }
  }
}

// ---------------- launcher ----------------

extern "C" void kernel_launch(void* const* d_in, const int* in_sizes, int n_in,
                              void* d_out, int out_size, void* d_ws, size_t ws_size,
                              hipStream_t stream) {
  const float* x    = (const float*)d_in[0];
  const float* mask = (const float*)d_in[1];
  const float* Wq   = (const float*)d_in[2];
  const float* bq   = (const float*)d_in[3];
  const float* Wk   = (const float*)d_in[4];
  const float* bk   = (const float*)d_in[5];
  const float* Wv   = (const float*)d_in[6];
  const float* bv   = (const float*)d_in[7];
  const float* Wo   = (const float*)d_in[8];
  const float* bo   = (const float*)d_in[9];

  char* ws = (char*)d_ws;
  const size_t MB = 1024 * 1024;
  __bf16* xb  = (__bf16*)(ws);             // x bf16           8 MB
  __bf16* Wqt = (__bf16*)(ws +  8 * MB);   // Wq^T bf16        2 MB
  __bf16* Wkt = (__bf16*)(ws + 10 * MB);
  __bf16* Wvt = (__bf16*)(ws + 12 * MB);
  __bf16* Wot = (__bf16*)(ws + 14 * MB);
  __bf16* Qb  = (__bf16*)(ws + 16 * MB);   // [b,h,t,d]        8 MB
  __bf16* Kb  = (__bf16*)(ws + 24 * MB);   // [b,h,t,d]        8 MB
  __bf16* Vb  = (__bf16*)(ws + 32 * MB);   // [b,h,d,t]        8 MB
  __bf16* Cb  = (__bf16*)(ws + 40 * MB);   // context bf16     8 MB

  cvt_bf16  <<<4096, 256, 0, stream>>>(x, xb, BATCH * SEQ * HID);
  cvt_bf16_t<<<4096, 256, 0, stream>>>(Wq, Wqt);
  cvt_bf16_t<<<4096, 256, 0, stream>>>(Wk, Wkt);
  cvt_bf16_t<<<4096, 256, 0, stream>>>(Wv, Wvt);
  cvt_bf16_t<<<4096, 256, 0, stream>>>(Wo, Wot);

  dim3 gg(HID / 128, (BATCH * SEQ) / 128);   // 8 x 32
  gemm_bf16<1><<<gg, 256, 0, stream>>>(xb, Wqt, bq, Qb);
  gemm_bf16<1><<<gg, 256, 0, stream>>>(xb, Wkt, bk, Kb);
  gemm_bf16<2><<<gg, 256, 0, stream>>>(xb, Wvt, bv, Vb);

  attn<<<BATCH * NH * 8, 256, 0, stream>>>(Qb, Kb, Vb, mask, Cb);

  gemm_bf16<0><<<gg, 256, 0, stream>>>(Cb, Wot, bo, d_out);
}
